// RoPEConformerMHSA_80994493268427
// MI455X (gfx1250) — matrix operand order repaired and checked by reading, not compiled, not run on hardware
//
#include <hip/hip_runtime.h>

#define DEV __device__ __forceinline__

typedef unsigned short u16;
typedef unsigned int   u32;
typedef unsigned long long u64;
typedef __attribute__((ext_vector_type(16))) __bf16 v16bf;
typedef __attribute__((ext_vector_type(8)))  float  v8f;
typedef __attribute__((ext_vector_type(4)))  u32    u32x4;
typedef __attribute__((ext_vector_type(8)))  u32    u32x8;
typedef __attribute__((ext_vector_type(2)))  u32    u32x2;
typedef __attribute__((ext_vector_type(4)))  float  f32x4;
typedef __attribute__((ext_vector_type(8)))  int    i32x8;
typedef __attribute__((ext_vector_type(4)))  int    i32x4;

constexpr int Tlen = 2048, Dm = 1024, Hn = 16, HD = 64, BT = 4096;

DEV u16 f2bf(float f){ u32 u = __float_as_uint(f); u32 r = u + 0x7FFFu + ((u >> 16) & 1u); return (u16)(r >> 16); }
DEV float bf2f(u16 h){ return __uint_as_float(((u32)h) << 16); }

DEV v16bf mkfrag(u32x4 lo, u32x4 hi){
  u32x8 t = {lo[0],lo[1],lo[2],lo[3],hi[0],hi[1],hi[2],hi[3]};
  return __builtin_bit_cast(v16bf, t);
}
DEV u32x4 g128(const u16* p){ return *(const u32x4*)p; }

DEV v8f vzero(){
  v8f z;
#pragma unroll
  for(int i=0;i<8;++i) z[i]=0.f;
  return z;
}
DEV v8f wmma_bf16(v16bf a, v16bf b, v8f c){
  return __builtin_amdgcn_wmma_f32_16x16x32_bf16(false, a, false, b, (short)0, c, false, false);
}

// LDS byte offset of a generic pointer into shared memory (LDS aperture keeps
// the LDS offset in addr[31:0]).
DEV u32 lds_off(const void* p){ return (u32)(size_t)p; }

// Per-lane async copy global -> LDS, 16 bytes, tracked by ASYNCcnt.
DEV void async_ld_b128(u32 ldsaddr, const void* gptr){
  asm volatile("global_load_async_to_lds_b128 %0, %1, off"
               :: "v"(ldsaddr), "v"(gptr) : "memory");
}
DEV void wait_asynccnt0(){ asm volatile("s_wait_asynccnt 0x0" ::: "memory"); }

// Tensor Data Mover: 2D tile (elements are 2 bytes) global -> LDS.
// D# packing per CDNA5 ISA.
DEV void tdm_load_2d(u32 ldsaddr, const void* gaddr,
                     u32 tdim0, u32 tdim1, u32 tile0, u32 tile1, u32 stride0){
  u64 ga = (u64)(size_t)gaddr;
  u32x4 g0;
  g0[0] = 1u;                              // count=1, user descriptor
  g0[1] = ldsaddr;                         // lds_addr
  g0[2] = (u32)ga;                         // global_addr[31:0]
  g0[3] = (u32)(ga >> 32) | (2u << 30);    // global_addr[56:32] | type=2
  i32x8 g1;
  g1[0] = (int)(1u << 16);                 // data_size=1 -> 2-byte elements
  g1[1] = (int)(tdim0 << 16);              // tensor_dim0[15:0]  (bits 63:48)
  g1[2] = (int)((tdim0 >> 16) | (tdim1 << 16)); // dim0 hi | dim1 lo
  g1[3] = (int)((tdim1 >> 16) | (tile0 << 16)); // dim1 hi | tile_dim0
  g1[4] = (int)tile1;                      // tile_dim1 (tile_dim2=0)
  g1[5] = (int)stride0;                    // tensor_dim0_stride[31:0]
  g1[6] = 0;
  g1[7] = 0;
  i32x4 z4; z4[0]=0; z4[1]=0; z4[2]=0; z4[3]=0;
#if __clang_major__ >= 23
  i32x8 z8; z8[0]=0; z8[1]=0; z8[2]=0; z8[3]=0; z8[4]=0; z8[5]=0; z8[6]=0; z8[7]=0;
  __builtin_amdgcn_tensor_load_to_lds(g0, g1, z4, z4, z8, 0);
#else
  __builtin_amdgcn_tensor_load_to_lds(g0, g1, z4, z4, 0);
#endif
}

// -------------------- fp32 -> bf16 weight convert --------------------
__global__ void cvt_kernel(const float* __restrict__ src, u16* __restrict__ dst){
  int idx = blockIdx.x*256 + threadIdx.x;
  f32x4 v = ((const f32x4*)src)[idx];
  u32x2 pk;
  pk[0] = (u32)f2bf(v[0]) | ((u32)f2bf(v[1])<<16);
  pk[1] = (u32)f2bf(v[2]) | ((u32)f2bf(v[3])<<16);
  ((u32x2*)dst)[idx] = pk;
}

// -------------------- LayerNorm + bf16 quantize --------------------
__global__ void ln_kernel(const float* __restrict__ x, const float* __restrict__ g,
                          const float* __restrict__ bet, u16* __restrict__ h){
  __shared__ float s1[256], s2[256];
  int row = blockIdx.x, tid = threadIdx.x;
  f32x4 v = ((const f32x4*)(x + (size_t)row*Dm))[tid];
  s1[tid] = v[0]+v[1]+v[2]+v[3];
  s2[tid] = v[0]*v[0]+v[1]*v[1]+v[2]*v[2]+v[3]*v[3];
  __syncthreads();
  for(int off=128; off>0; off>>=1){
    if(tid<off){ s1[tid]+=s1[tid+off]; s2[tid]+=s2[tid+off]; }
    __syncthreads();
  }
  float mu  = s1[0]*(1.f/Dm);
  float var = s2[0]*(1.f/Dm) - mu*mu;
  float rs  = rsqrtf(var + 1e-5f);
  int c = tid*4;
  u32x2 pk;
  u16 h0 = f2bf((v[0]-mu)*rs*g[c+0] + bet[c+0]);
  u16 h1 = f2bf((v[1]-mu)*rs*g[c+1] + bet[c+1]);
  u16 h2 = f2bf((v[2]-mu)*rs*g[c+2] + bet[c+2]);
  u16 h3 = f2bf((v[3]-mu)*rs*g[c+3] + bet[c+3]);
  pk[0] = (u32)h0 | ((u32)h1<<16);
  pk[1] = (u32)h2 | ((u32)h3<<16);
  ((u32x2*)(h + (size_t)row*Dm))[tid] = pk;
}

// -------------------- WMMA GEMM: out[m,n] = sum_k A[m,k]*W[n,k] --------------------
enum { OUT_BF16_ROW = 0, OUT_BF16_VT = 1, OUT_F32_ROW = 2 };

__global__ void gemm_kernel(const u16* __restrict__ A, const u16* __restrict__ W,
                            void* __restrict__ out, int mode){
  __shared__ __align__(16) u16 Bs[64*32];   // [n_local 0..63][k_local 0..31]
  const int tid  = threadIdx.x;
  const int wave = tid >> 5, lane = tid & 31;
  const int ln16 = lane & 15, hi = lane >> 4;
  const int row0 = blockIdx.x*128 + wave*16;
  const int col0 = blockIdx.y*64;
  const u16* arow = A + (size_t)(row0 + ln16)*Dm;
  const int ac0 = hi ? 16 : 0;             // A-frag K-chunk start for this lane half

  v8f acc[4];
#pragma unroll
  for(int j=0;j<4;++j) acc[j] = vzero();

  const int nl = tid >> 2, ch = tid & 3;
  const u32 bsdst = lds_off(&Bs[nl*32 + ch*8]);

  for(int kk=0; kk<Dm; kk+=32){
    __syncthreads();
    // async stage of 64x32 weight tile into LDS (256 x async b128, ASYNCcnt)
    async_ld_b128(bsdst, W + (size_t)(col0+nl)*Dm + kk + ch*8);
    wait_asynccnt0();
    __syncthreads();
    if(kk + 32 < Dm) __builtin_prefetch(arow + kk + 32, 0, 1);

    v16bf af = mkfrag(g128(arow + kk + ac0), g128(arow + kk + ac0 + 8));
#pragma unroll
    for(int j=0;j<4;++j){
      const u16* bp = &Bs[(j*16 + ln16)*32 + hi*16];
      v16bf bf = mkfrag(*(const u32x4*)bp, *(const u32x4*)(bp+8));
      acc[j] = wmma_bf16(af, bf, acc[j]);
    }
  }

  const int mrow = row0 + hi*8;
#pragma unroll
  for(int j=0;j<4;++j){
#pragma unroll
    for(int r=0;r<8;++r){
      int m = mrow + r;
      int n = col0 + j*16 + ln16;
      float val = acc[j][r];
      if(mode == OUT_F32_ROW){
        ((float*)out)[(size_t)m*Dm + n] = val;
      } else if(mode == OUT_BF16_ROW){
        ((u16*)out)[(size_t)m*Dm + n] = f2bf(val);
      } else { // V transposed: [b][h][d][t]
        int b = m >> 11, t = m & (Tlen-1);
        int hh = n >> 6, d = n & 63;
        ((u16*)out)[(((size_t)(b*Hn + hh)*HD + d) << 11) + t] = f2bf(val);
      }
    }
  }
}

// -------------------- RoPE (in place on bf16 Q or K) --------------------
__global__ void rope_kernel(u16* __restrict__ x, const float* __restrict__ cs,
                            const float* __restrict__ sn){
  int idx = blockIdx.x*256 + threadIdx.x;   // BT*Hn*32 pairs total
  int row = idx >> 9;                       // 512 pairs per row
  int pr  = idx & 511;
  int hh  = pr >> 5;
  int i   = pr & 31;
  int t   = row & (Tlen-1);
  size_t base = (size_t)row*Dm + hh*HD + 2*i;
  float x0 = bf2f(x[base]), x1 = bf2f(x[base+1]);
  float c = cs[t*32 + i], s = sn[t*32 + i];
  x[base]   = f2bf(x0*c - x1*s);
  x[base+1] = f2bf(x0*s + x1*c);
}

// -------------------- Flash attention (non-causal), TDM-staged K/V --------------------
__global__ void attn_kernel(const u16* __restrict__ q, const u16* __restrict__ k,
                            const u16* __restrict__ vt, u16* __restrict__ outp){
  __shared__ __align__(16) u16 Ks[32*64];    // [key 0..31][d 0..63]
  __shared__ __align__(16) u16 Vs[64*32];    // [d 0..63][key 0..31]
  __shared__ __align__(16) u16 P[8][16*32];  // per-wave P transpose staging
  const int tid  = threadIdx.x;
  const int wave = tid >> 5, lane = tid & 31;
  const int ln16 = lane & 15, hi = lane >> 4;
  const int bh = blockIdx.y, b = bh >> 4, hh = bh & 15;
  const int q0 = blockIdx.x*128 + wave*16;
  const int ac0 = hi ? 16 : 0;

  const u16* qrow = q + (size_t)(b*Tlen + q0 + ln16)*Dm + hh*HD;
  v16bf a0 = mkfrag(g128(qrow + ac0),      g128(qrow + ac0 + 8));       // d 0..31
  v16bf a1 = mkfrag(g128(qrow + 32 + ac0), g128(qrow + 32 + ac0 + 8));  // d 32..63

  v8f oacc[4];
#pragma unroll
  for(int j=0;j<4;++j) oacc[j] = vzero();
  float rmax[8], rsum[8];
#pragma unroll
  for(int r=0;r<8;++r){ rmax[r] = -1e30f; rsum[r] = 0.f; }

  u16* pl = P[wave];
  const u16* kbh = k  + (size_t)(b*Tlen)*Dm + hh*HD;   // K rows for this (b,h)
  const u16* vbh = vt + (size_t)bh*HD*Tlen;            // V^T rows for this (b,h)
  const u32 ksoff = lds_off(&Ks[0]);
  const u32 vsoff = lds_off(&Vs[0]);

  for(int kb=0; kb<Tlen; kb+=32){
    __syncthreads();                       // previous tile fully consumed
    if(wave == 0){
      // K tile: 32 rows (keys) x 64 elems (d), row stride Dm
      tdm_load_2d(ksoff, kbh + (size_t)kb*Dm, HD, 32, HD, 32, Dm);
      // V^T tile: 64 rows (d) x 32 elems (keys), row stride Tlen
      tdm_load_2d(vsoff, vbh + kb, 32, HD, 32, HD, Tlen);
      __builtin_amdgcn_s_wait_tensorcnt(0);
    }
    __syncthreads();

    // S = Q * K^T for 32 keys (two 16x16 tiles, contraction over hd=64)
    v8f s0 = vzero(), s1 = vzero();
#pragma unroll
    for(int g=0; g<2; ++g){
      const u16* kr = &Ks[(g*16 + ln16)*HD + hi*16];
      v16bf b0 = mkfrag(*(const u32x4*)kr,      *(const u32x4*)(kr+8));   // d 0..31
      v16bf b1 = mkfrag(*(const u32x4*)(kr+32), *(const u32x4*)(kr+40));  // d 32..63
      v8f sg = (g==0) ? s0 : s1;
      sg = wmma_bf16(a0, b0, sg);
      sg = wmma_bf16(a1, b1, sg);
      if(g==0) s0 = sg; else s1 = sg;
    }

    // online softmax over the 32 new keys (row-wise across 16 lanes)
#pragma unroll
    for(int r=0;r<8;++r){
      float v0 = s0[r]*0.125f, v1 = s1[r]*0.125f;  // 1/sqrt(64)
      float mx = fmaxf(v0, v1);
#pragma unroll
      for(int off=1; off<16; off<<=1) mx = fmaxf(mx, __shfl_xor(mx, off, 32));
      float nm   = fmaxf(rmax[r], mx);
      float corr = __expf(rmax[r] - nm);
      rmax[r] = nm;
      float p0 = __expf(v0 - nm), p1 = __expf(v1 - nm);
      float ps = p0 + p1;
#pragma unroll
      for(int off=1; off<16; off<<=1) ps += __shfl_xor(ps, off, 32);
      rsum[r] = rsum[r]*corr + ps;
      oacc[0][r]*=corr; oacc[1][r]*=corr; oacc[2][r]*=corr; oacc[3][r]*=corr;
      int m = r + hi*8;
      pl[m*32 + ln16]      = f2bf(p0);
      pl[m*32 + 16 + ln16] = f2bf(p1);
    }

    // P (16x32) as A-fragment via LDS transpose
  v16bf pa = mkfrag(*(const u32x4*)&pl[ln16*32 + ac0],
                      *(const u32x4*)&pl[ln16*32 + ac0 + 8]);

    // O += P * V   (V^T tile in LDS: contiguous along key for fixed d)
#pragma unroll
    for(int j=0;j<4;++j){
      const u16* vp = &Vs[(j*16 + ln16)*32 + hi*16];
      v16bf vb = mkfrag(*(const u32x4*)vp, *(const u32x4*)(vp+8));
      oacc[j] = wmma_bf16(pa, vb, oacc[j]);
    }
  }

#pragma unroll
  for(int r=0;r<8;++r) rsum[r] = 1.f / rsum[r];
#pragma unroll
  for(int j=0;j<4;++j){
#pragma unroll
    for(int r=0;r<8;++r){
      int m   = q0 + r + hi*8;
      int col = hh*HD + j*16 + ln16;
      outp[(size_t)(b*Tlen + m)*Dm + col] = f2bf(oacc[j][r]*rsum[r]);
    }
  }
}

// -------------------- launch --------------------
extern "C" void kernel_launch(void* const* d_in, const int* in_sizes, int n_in,
                              void* d_out, int out_size, void* d_ws, size_t ws_size,
                              hipStream_t stream){
  (void)in_sizes; (void)n_in; (void)out_size; (void)ws_size;
  const float* x  = (const float*)d_in[0];
  const float* wq = (const float*)d_in[1];
  const float* wk = (const float*)d_in[2];
  const float* wv = (const float*)d_in[3];
  const float* wo = (const float*)d_in[4];
  const float* lg = (const float*)d_in[5];
  const float* lb = (const float*)d_in[6];
  const float* cs = (const float*)d_in[7];
  const float* sn = (const float*)d_in[8];

  u16* ws   = (u16*)d_ws;
  u16* wqb  = ws;
  u16* wkb  = wqb + (size_t)Dm*Dm;
  u16* wvb  = wkb + (size_t)Dm*Dm;
  u16* wob  = wvb + (size_t)Dm*Dm;
  u16* hb   = wob + (size_t)Dm*Dm;
  u16* qb   = hb   + (size_t)BT*Dm;
  u16* kbuf = qb   + (size_t)BT*Dm;
  u16* vtb  = kbuf + (size_t)BT*Dm;
  u16* ob   = vtb  + (size_t)BT*Dm;

  cvt_kernel<<<1024, 256, 0, stream>>>(wq, wqb);
  cvt_kernel<<<1024, 256, 0, stream>>>(wk, wkb);
  cvt_kernel<<<1024, 256, 0, stream>>>(wv, wvb);
  cvt_kernel<<<1024, 256, 0, stream>>>(wo, wob);

  ln_kernel<<<4096, 256, 0, stream>>>(x, lg, lb, hb);

  gemm_kernel<<<dim3(32,16), 256, 0, stream>>>(hb, wqb, qb,   OUT_BF16_ROW);
  gemm_kernel<<<dim3(32,16), 256, 0, stream>>>(hb, wkb, kbuf, OUT_BF16_ROW);
  gemm_kernel<<<dim3(32,16), 256, 0, stream>>>(hb, wvb, vtb,  OUT_BF16_VT);

  rope_kernel<<<8192, 256, 0, stream>>>(qb,   cs, sn);
  rope_kernel<<<8192, 256, 0, stream>>>(kbuf, cs, sn);

  attn_kernel<<<dim3(16,32), 256, 0, stream>>>(qb, kbuf, vtb, ob);

  gemm_kernel<<<dim3(32,16), 256, 0, stream>>>(ob, wob, d_out, OUT_F32_ROW);
}
